// NCA3D_36060545417696
// MI455X (gfx1250) — compile-verified, matrix operand order checked by reading, not run on hardware
//
#include <hip/hip_runtime.h>
#include <hip/hip_bf16.h>

// ---------------- types ----------------
typedef __bf16 bf16_t;
typedef __attribute__((ext_vector_type(16))) __bf16 v16bf;
typedef __attribute__((ext_vector_type(8)))  __bf16 v8bf;
typedef __attribute__((ext_vector_type(8)))  float  v8f;

#define C_    32
#define HID_  128
#define DHW   262144         // 64^3
#define HW_   4096           // 64*64
#define NVOX  524288u        // 2 * 64^3
#define NROWS 8192           // 2*64*64 (b,d,h) rows

// total dynamic LDS (bytes): 22144 floats + 26880 bf16
#define SMEM_BYTES (22144*4 + 26880*2)

// CDNA5 async global->LDS copies (ASYNCcnt-tracked). Set to 0 to fall back to
// plain load+store staging if the asm mnemonic is rejected.
#define USE_ASYNC_LDS 1

// ---------------- helpers ----------------
__host__ __device__ __forceinline__ void threefry2x32(unsigned k0, unsigned k1,
                                                      unsigned x0, unsigned x1,
                                                      unsigned& o0, unsigned& o1) {
  const unsigned ks[3] = { k0, k1, k0 ^ k1 ^ 0x1BD11BDAu };
  const int R[8] = {13,15,26,6,17,29,16,24};
  x0 += ks[0]; x1 += ks[1];
  #pragma unroll
  for (int g = 0; g < 5; ++g) {
    #pragma unroll
    for (int r = 0; r < 4; ++r) {
      int rot = R[(g & 1) * 4 + r];
      x0 += x1;
      x1 = (x1 << rot) | (x1 >> (32 - rot));
      x1 ^= x0;
    }
    x0 += ks[(g + 1) % 3];
    x1 += ks[(g + 2) % 3] + (unsigned)(g + 1);
  }
  o0 = x0; o1 = x1;
}

__device__ __forceinline__ float jax_mask(unsigned k0, unsigned k1, unsigned i) {
  // JAX counter-mode: counts = iota(N); halves (i, i+N/2)
  const unsigned Nh = NVOX / 2u;
  unsigned o0, o1, bits;
  if (i < Nh) { threefry2x32(k0, k1, i, i + Nh, o0, o1); bits = o0; }
  else        { threefry2x32(k0, k1, i - Nh, i, o0, o1); bits = o1; }
  float u = __builtin_bit_cast(float, (bits >> 9) | 0x3F800000u) - 1.0f;
  return (u > 0.5f) ? 1.0f : 0.0f;
}

__host__ __device__ __forceinline__ unsigned short bf_bits(float f) { // RNE
  unsigned u = __builtin_bit_cast(unsigned, f);
  return (unsigned short)((u + 0x7FFFu + ((u >> 16) & 1u)) >> 16);
}

__device__ __forceinline__ bf16_t f2bf(float f) {
  unsigned short s = bf_bits(f);
  return __builtin_bit_cast(bf16_t, s);
}

__device__ __forceinline__ int refl(int i) {          // reflect pad, size 64
  return i < 0 ? -i : (i > 63 ? 126 - i : i);
}

// 16B global -> LDS copy. Async (TDM-adjacent path, ASYNCcnt) when enabled.
__device__ __forceinline__ void cp16_g2s(void* lds_dst, const void* gsrc) {
#if USE_ASYNC_LDS
  unsigned lds_off = (unsigned)(unsigned long long)lds_dst; // [31:0] = LDS addr
  asm volatile("global_load_async_to_lds_b128 %0, %1, off"
               :: "v"(lds_off), "v"(gsrc) : "memory");
#else
  *(float4*)lds_dst = *(const float4*)gsrc;
#endif
}

__device__ __forceinline__ void g2s_wait() {
#if USE_ASYNC_LDS
#if __has_builtin(__builtin_amdgcn_s_wait_asynccnt)
  __builtin_amdgcn_s_wait_asynccnt(0);
#else
  asm volatile("s_wait_asynccnt 0" ::: "memory");
#endif
#endif
}

// assemble a 16-bit A/B WMMA fragment: two 16B LDS loads per lane
// lane(0-15):  e0..7 -> K = k0+8*half+e ; e8..15 -> K = k0+16+8*half+(e-8)
__device__ __forceinline__ v16bf load_frag(const bf16_t* row, int half, int k0) {
  v8bf lo = *(const v8bf*)(row + k0 + 8 * half);
  v8bf hi = *(const v8bf*)(row + k0 + 16 + 8 * half);
  return __builtin_shufflevector(lo, hi, 0,1,2,3,4,5,6,7,8,9,10,11,12,13,14,15);
}

__device__ __forceinline__ v8f wmma_bf16(v16bf a, v16bf b, v8f c) {
  return __builtin_amdgcn_wmma_f32_16x16x32_bf16(false, a, false, b,
                                                 (short)0, c, false, false);
}

// ---------------- fused step kernel (STATS pass / UPDATE pass) --------------
template <bool STATS>
__global__ __launch_bounds__(128)
void nca_step_kernel(const float* __restrict__ state,      // [2,32,64,64,64] cur
                     float*       __restrict__ state_next, // next (UPDATE only)
                     const float* __restrict__ conv_w,     // [32,1,3,3,3]
                     const float* __restrict__ conv_b,     // [32]
                     const unsigned short* __restrict__ fc0wbf, // [128,64] bf16
                     const unsigned short* __restrict__ fc1wbf, // [32,128] bf16 (row31=0)
                     const float* __restrict__ fc0_b,      // [128]
                     const float* __restrict__ scale,      // [128] (UPDATE)
                     const float* __restrict__ shift2,     // [128] (UPDATE)
                     float* __restrict__ gsum,             // [128] (STATS)
                     float* __restrict__ gsq,              // [128] (STATS)
                     unsigned key0, unsigned key1)
{
  extern __shared__ char smem_raw[];
  float*  halo   = (float*)smem_raw;           // 32*3*3*64 = 18432 f
  float*  convw  = halo   + 18432;             // 864
  float*  convb  = convw  + 864;               // 32
  float*  scl    = convb  + 32;                // 128
  float*  sh2    = scl    + 128;               // 128
  float*  fc0b   = sh2    + 128;               // 128
  float*  sums   = fc0b   + 128;               // 128
  float*  sumsq  = sums   + 128;               // 128
  float*  maskf  = sumsq  + 128;               // 64
  float*  deltaS = maskf  + 64;                // 64*33 = 2112
  bf16_t* fc0w   = (bf16_t*)(deltaS + 2112);   // 128 rows x 72 (pad)
  bf16_t* fc1w   = fc0w + 128 * 72;            // 32 rows x 136 (pad)
  bf16_t* catl   = fc1w + 32 * 136;            // 64 vox x 72 (pad)
  bf16_t* hA     = catl + 64 * 72;             // 4 waves x 16 x 136 (pad)

  const int t  = threadIdx.x;
  const int wg = blockIdx.x;                   // row id over (b,d,h)
  const int b  = wg >> 12;
  const int d0 = (wg >> 6) & 63;
  const int h0 = wg & 63;
  const float* sb = state + (size_t)b * C_ * DHW;

  // ---- stage conv halo via async b128 copies: [c][dd][hh][w] (reflect d/h)
  for (int ch = t; ch < 32 * 3 * 3 * 16; ch += 128) {   // 4608 x 16B chunks
    const int e  = ch * 4;
    const int w  = e & 63;
    const int hh = (e >> 6) % 3;
    const int dd = ((e >> 6) / 3) % 3;
    const int c  = e / 576;
    const int dr = refl(d0 - 1 + dd), hr = refl(h0 - 1 + hh);
    cp16_g2s(halo + e, sb + (size_t)c * DHW + (size_t)dr * HW_ + hr * 64 + w);
  }
  // ---- stage fc0 weights (pre-converted bf16): rows of 64 -> padded 72
  for (int ch = t; ch < HID_ * 8; ch += 128) {          // 1024 x 16B chunks
    const int n = ch >> 3, k = (ch & 7) * 8;
    cp16_g2s(fc0w + n * 72 + k, fc0wbf + n * 64 + k);
  }
  if (!STATS) {
    // fc1 weights (row 31 pre-zeroed): rows of 128 -> padded 136
    for (int ch = t; ch < 32 * 16; ch += 128) {         // 512 x 16B chunks
      const int n = ch >> 4, k = (ch & 15) * 8;
      cp16_g2s(fc1w + n * 136 + k, fc1wbf + n * 128 + k);
    }
  }
  // ---- small tables (plain loads)
  for (int e = t; e < C_ * 27; e += 128) convw[e] = conv_w[e];
  if (t < 32) convb[t] = conv_b[t];
  if (STATS) {
    if (t < 128) { fc0b[t] = fc0_b[t]; sums[t] = 0.f; sumsq[t] = 0.f; }
  } else {
    if (t < 128) { scl[t] = scale[t]; sh2[t] = shift2[t]; }
    if (t < 64) {
      unsigned i = (unsigned)((((b * 64 + d0) * 64 + h0) * 64) + t);
      maskf[t] = jax_mask(key0, key1, i);
    }
  }
  g2s_wait();            // drain ASYNCcnt for this wave's transfers
  __syncthreads();

  // ---- build cat tile: 64 voxels x (32 state + 32 depthwise-conv), bf16
  for (int e = t; e < 64 * 32; e += 128) {
    int v = e & 63;          // voxel == w coordinate
    int c = e >> 6;
    catl[v * 72 + c] = f2bf(halo[(c * 9 + 4) * 64 + v]);   // center value
    float acc = convb[c];
    const float* wc = convw + c * 27;
    const float* hc = halo + c * 9 * 64;
    const int wm1 = (v == 0) ? 1 : v - 1;
    const int wp1 = (v == 63) ? 62 : v + 1;
    #pragma unroll
    for (int dd = 0; dd < 3; ++dd)
      #pragma unroll
      for (int hh = 0; hh < 3; ++hh) {
        const float* hr = hc + (dd * 3 + hh) * 64;
        const float* wr = wc + (dd * 3 + hh) * 3;
        acc += wr[0] * hr[wm1] + wr[1] * hr[v] + wr[2] * hr[wp1];
      }
    catl[v * 72 + 32 + c] = f2bf(acc);
  }
  __syncthreads();

  // ---- fc0 via WMMA: per wave, 16 voxels, K=64, N=128 (8 tiles x 2 K-chunks)
  const int wave = t >> 5;
  const int lane = t & 31;
  const int half = lane >> 4;
  const int mr   = lane & 15;

  const bf16_t* arow = catl + (wave * 16 + mr) * 72;
  v16bf a0 = load_frag(arow, half, 0);
  v16bf a1 = load_frag(arow, half, 32);

  v8f acc[8];
  #pragma unroll
  for (int nt = 0; nt < 8; ++nt) {
    const bf16_t* brow = fc0w + (nt * 16 + mr) * 72;
    v16bf b0 = load_frag(brow, half, 0);
    v16bf b1 = load_frag(brow, half, 32);
    v8f c = {};
    c = wmma_bf16(a0, b0, c);
    c = wmma_bf16(a1, b1, c);
    acc[nt] = c;
  }

  if (STATS) {
    // ---- accumulate per-hidden-channel sum / sum-of-squares (bias included)
    #pragma unroll
    for (int nt = 0; nt < 8; ++nt) {
      const int n = nt * 16 + mr;
      const float bias = fc0b[n];
      float s = 0.f, q = 0.f;
      #pragma unroll
      for (int j = 0; j < 8; ++j) {
        float hv = acc[nt][j] + bias;
        s += hv; q += hv * hv;
      }
      s += __shfl_xor(s, 16);
      q += __shfl_xor(q, 16);
      if (half == 0) { atomicAdd(&sums[n], s); atomicAdd(&sumsq[n], q); }
    }
    __syncthreads();
    if (t < 128) { atomicAdd(&gsum[t], sums[t]); atomicAdd(&gsq[t], sumsq[t]); }
  } else {
    // ---- BN (folded scale/shift incl. fc0 bias) + ReLU -> hA (A layout via LDS)
    bf16_t* hbase = hA + wave * 16 * 136;
    #pragma unroll
    for (int nt = 0; nt < 8; ++nt) {
      const int n = nt * 16 + mr;
      const float s = scl[n], z = sh2[n];
      #pragma unroll
      for (int j = 0; j < 8; ++j) {
        float hv = acc[nt][j] * s + z;
        hv = hv > 0.f ? hv : 0.f;
        hbase[(j + 8 * half) * 136 + n] = f2bf(hv);
      }
    }
    __syncthreads();

    // ---- fc1 via WMMA: K=128, N=32 (31 real + 1 zero pad)
    const bf16_t* ha_row = hbase + mr * 136;
    v16bf ha0 = load_frag(ha_row, half, 0);
    v16bf ha1 = load_frag(ha_row, half, 32);
    v16bf ha2 = load_frag(ha_row, half, 64);
    v16bf ha3 = load_frag(ha_row, half, 96);
    #pragma unroll
    for (int nt = 0; nt < 2; ++nt) {
      const bf16_t* brow = fc1w + (nt * 16 + mr) * 136;
      v8f dacc = {};
      dacc = wmma_bf16(ha0, load_frag(brow, half, 0),  dacc);
      dacc = wmma_bf16(ha1, load_frag(brow, half, 32), dacc);
      dacc = wmma_bf16(ha2, load_frag(brow, half, 64), dacc);
      dacc = wmma_bf16(ha3, load_frag(brow, half, 96), dacc);
      const int o = nt * 16 + mr;
      #pragma unroll
      for (int j = 0; j < 8; ++j) {
        const int v = wave * 16 + j + 8 * half;
        if (o < 31) deltaS[v * 33 + o] = dacc[j];
      }
    }
    __syncthreads();

    // ---- residual + fire mask, coalesced channel-row stores
    for (int e = t; e < 31 * 64; e += 128) {
      const int w = e & 63;
      const int o = e >> 6;
      const int c = 1 + o;
      const float val = halo[(c * 9 + 4) * 64 + w] + deltaS[w * 33 + o] * maskf[w];
      state_next[((size_t)(b * C_ + c)) * DHW + (size_t)d0 * HW_ + h0 * 64 + w] = val;
    }
  }
}

// ---------------- small kernels ----------------
__global__ void prep_weights(const float* __restrict__ fc0_w,
                             const float* __restrict__ fc1_w,
                             unsigned short* __restrict__ fc0wbf,
                             unsigned short* __restrict__ fc1wbf) {
  const int i = blockIdx.x * blockDim.x + threadIdx.x;
  if (i < HID_ * 64) fc0wbf[i] = bf_bits(fc0_w[i]);
  const int j = i - HID_ * 64;
  if (j >= 0 && j < 32 * 128) {
    const int n = j >> 7, k = j & 127;
    fc1wbf[j] = (n < 31) ? bf_bits(fc1_w[n * 128 + k]) : (unsigned short)0;
  }
}

__global__ void bn_finalize(float* gsum, float* gsq,
                            const float* __restrict__ bn_w,
                            const float* __restrict__ bn_b,
                            const float* __restrict__ fc0_b,
                            float* scale, float* shift2) {
  const int t = threadIdx.x;
  if (t < 128) {
    const float inv_n = 1.0f / (float)NVOX;
    float mean = gsum[t] * inv_n;
    float var  = gsq[t] * inv_n - mean * mean;
    float inv  = rsqrtf(var + 1e-5f);
    float sc   = bn_w[t] * inv;
    scale[t]  = sc;
    shift2[t] = bn_b[t] - mean * sc + fc0_b[t] * sc;  // fc0 bias folded in
    gsum[t] = 0.f; gsq[t] = 0.f;                      // re-arm accumulators
  }
}

__global__ void nca_init(const float* __restrict__ x,
                         float* sA, float* sB, float* gsum, float* gsq) {
  const size_t i = (size_t)blockIdx.x * blockDim.x + threadIdx.x;
  if (i < 128) { gsum[i] = 0.f; gsq[i] = 0.f; }
  const size_t total = (size_t)2 * C_ * DHW;
  if (i >= total) return;
  const size_t c = (i / DHW) % C_;
  const size_t b = i / ((size_t)C_ * DHW);
  const size_t r = i % DHW;
  if (c == 0) { float v = x[b * DHW + r]; sA[i] = v; sB[i] = v; }
  else        { sA[i] = 0.f; }
}

__global__ void nca_out(const float* __restrict__ s, float* __restrict__ out) {
  const size_t i = (size_t)blockIdx.x * blockDim.x + threadIdx.x;
  const size_t total = (size_t)2 * 3 * DHW;
  if (i >= total) return;
  const size_t b  = i / ((size_t)3 * DHW);
  const size_t cc = (i / DHW) % 3;
  const size_t r  = i % DHW;
  out[i] = s[(b * C_ + 1 + cc) * (size_t)DHW + r];
}

// ---------------- host launcher ----------------
extern "C" void kernel_launch(void* const* d_in, const int* in_sizes, int n_in,
                              void* d_out, int out_size, void* d_ws, size_t ws_size,
                              hipStream_t stream) {
  const float* x      = (const float*)d_in[0];
  const float* conv_w = (const float*)d_in[1];
  const float* conv_b = (const float*)d_in[2];
  const float* fc0_w  = (const float*)d_in[3];
  const float* fc0_b  = (const float*)d_in[4];
  const float* fc1_w  = (const float*)d_in[5];
  const float* bn_w   = (const float*)d_in[6];
  const float* bn_b   = (const float*)d_in[7];
  float* out = (float*)d_out;

  float* ws = (float*)d_ws;
  const size_t state_elems = (size_t)2 * C_ * DHW;   // 16.78M floats
  float* sA     = ws;
  float* sB     = sA + state_elems;
  float* gsum   = sB + state_elems;
  float* gsq    = gsum + 128;
  float* scale  = gsq + 128;
  float* shift2 = scale + 128;
  unsigned short* fc0wbf = (unsigned short*)(shift2 + 128); // 8192 bf16 (16B-aligned)
  unsigned short* fc1wbf = fc0wbf + HID_ * 64;              // 4096 bf16

  // init: state buffers, stat accumulators, bf16 weight images
  nca_init<<<(unsigned)((state_elems + 255) / 256), 256, 0, stream>>>(x, sA, sB, gsum, gsq);
  prep_weights<<<48, 256, 0, stream>>>(fc0_w, fc1_w, fc0wbf, fc1wbf);

  for (int step = 0; step < 10; ++step) {
    unsigned nk0, nk1;                       // jax fold_in(key(42), step)
    threefry2x32(0u, 42u, 0u, (unsigned)step, nk0, nk1);
    const float* cur = (step & 1) ? sB : sA;
    float*       nxt = (step & 1) ? sA : sB;

    nca_step_kernel<true><<<NROWS, 128, SMEM_BYTES, stream>>>(
        cur, nullptr, conv_w, conv_b, fc0wbf, fc1wbf, fc0_b,
        nullptr, nullptr, gsum, gsq, 0u, 0u);

    bn_finalize<<<1, 128, 0, stream>>>(gsum, gsq, bn_w, bn_b, fc0_b, scale, shift2);

    nca_step_kernel<false><<<NROWS, 128, SMEM_BYTES, stream>>>(
        cur, nxt, conv_w, conv_b, fc0wbf, fc1wbf, fc0_b,
        scale, shift2, nullptr, nullptr, nk0, nk1);
  }

  // after step 9 (odd), final state is sA; emit channels 1..3
  nca_out<<<(unsigned)((2 * 3 * DHW + 255) / 256), 256, 0, stream>>>(sA, out);
}
